// base_fcrn_13640816132229
// MI455X (gfx1250) — matrix operand order repaired and checked by reading, not compiled, not run on hardware
//
#include <hip/hip_runtime.h>
#include <cstdint>

typedef __bf16 bf16_t;
typedef bf16_t v16bf __attribute__((ext_vector_type(16)));
typedef bf16_t v8bf  __attribute__((ext_vector_type(8)));
typedef bf16_t v4bf  __attribute__((ext_vector_type(4)));
typedef float  v8f   __attribute__((ext_vector_type(8)));

#define LDS_RS 40   // A-tile row stride in halves: 80B -> conflict-free & 16B aligned

// ---------------------------------------------------------------------------
// Weight pack: OIHW fp32 -> [tap][kb][coutP][k32] bf16, couts padded to 64.
// A lane's WMMA B-fragment (K = hi*8..+7, 16+hi*8..+7 of one cout) is two 16B runs in
// memory -> two global_load_b128, no guards in the GEMM hot loop.
// ---------------------------------------------------------------------------
__global__ __launch_bounds__(256) void pack_w_kernel(
    const float* __restrict__ w, bf16_t* __restrict__ wp,
    int Cout, int CoutP, int Cin, int Kh, int Kw, long long total)
{
  long long i = (long long)blockIdx.x * 256 + threadIdx.x;
  if (i >= total) return;
  int k = (int)(i & 31);
  long long r = i >> 5;
  int n = (int)(r % CoutP);
  long long r2 = r / CoutP;
  int KB  = Cin >> 5;
  int kb  = (int)(r2 % KB);
  int tap = (int)(r2 / KB);
  int kw = tap % Kw, kh = tap / Kw;
  int c = kb * 32 + k;
  bf16_t v = (bf16_t)0.0f;
  if (n < Cout) v = (bf16_t)w[(((long long)n * Cin + c) * Kh + kh) * Kw + kw];
  wp[i] = v;
}

__global__ __launch_bounds__(256) void fill_zero_kernel(float* __restrict__ p, long long n)
{
  long long i = (long long)blockIdx.x * 256 + threadIdx.x;
  if (i < n) p[i] = 0.0f;
}

__global__ __launch_bounds__(256) void residual_relu_kernel(
    float* __restrict__ a, const float* __restrict__ b, long long n)
{
  long long i = (long long)blockIdx.x * 256 + threadIdx.x;
  if (i < n) a[i] = fmaxf(a[i] + b[i], 0.0f);
}

__global__ __launch_bounds__(256) void upsample2x_kernel(
    const float* __restrict__ in, float* __restrict__ out, int NC, int H, int W)
{
  long long total = (long long)NC * 4 * H * W;
  long long i = (long long)blockIdx.x * 256 + threadIdx.x;
  if (i >= total) return;
  int W2 = 2 * W, H2 = 2 * H;
  int c = (int)(i % W2);
  long long r0 = i / W2;
  int r = (int)(r0 % H2);
  long long b = r0 / H2;
  out[i] = in[(b * H + (r >> 1)) * W + (c >> 1)];
}

// ---------------------------------------------------------------------------
// Implicit-GEMM conv: M = N*Ho*Wo pixels, Ndim = cout, K = Cin*Kh*Kw.
// Block = 128 thr = 4 waves, block tile 64(M) x 64(N).
// Wave (wm=w&1, wn=w>>1) owns a 32x32 sub-tile -> 2x2 WMMA accumulators:
// 16 v_wmma_f32_16x16x32_bf16 per block K-step, one shared LDS A tile.
// ---------------------------------------------------------------------------
__global__ __launch_bounds__(128) void conv_wmma_kernel(
    const float* __restrict__ x, const bf16_t* __restrict__ wp,
    const float* __restrict__ bias, float* __restrict__ y,
    int N, int Cin, int H, int W,
    int Cout, int CoutP, int Kh, int Kw, int ph, int pw,
    int Ho, int Wo,
    int Yh, int Yw, int rstep, int roff, int cstep, int coff,
    int reluIn, int reluOut)
{
  __shared__ __align__(16) bf16_t As[64 * LDS_RS];   // [m][k], padded rows
  const int tid  = threadIdx.x;
  const int lane = tid & 31;
  const int wave = tid >> 5;
  const int hi   = lane >> 4;       // lane-half -> K-group select
  const int ml   = lane & 15;
  const int wm   = wave & 1;        // M half of block tile
  const int wn   = wave >> 1;       // N half of block tile
  const int m0   = blockIdx.x * 64;
  const int HoWo = Ho * Wo;
  const int M    = N * HoWo;
  const int HW   = H * W;
  const int KB   = Cin >> 5;

  // Cooperative A-load: thread covers cin k = k4*4..+3 and pixels
  // m = (tid&15) + e*16 (adjacent lanes -> adjacent rows: conflict-free
  // ds_store_b64; fixed-k planes read 16-pixel coalesced per lane group).
  const int k4 = tid >> 4;          // 0..7
  const int mr = tid & 15;          // row within 16-group

  int  pm_oh[4], pm_ow[4];
  long long pm_base[4];
  bool pm_v[4];
#pragma unroll
  for (int e = 0; e < 4; ++e) {
    int mg = m0 + mr + e * 16;
    pm_v[e] = (mg < M);
    int mgc = pm_v[e] ? mg : 0;
    int n  = mgc / HoWo;
    int r  = mgc - n * HoWo;
    pm_oh[e] = r / Wo;
    pm_ow[e] = r - pm_oh[e] * Wo;
    pm_base[e] = (long long)n * Cin * HW;
  }

  v8f acc[2][2];
#pragma unroll
  for (int mt = 0; mt < 2; ++mt)
#pragma unroll
    for (int nt = 0; nt < 2; ++nt)
#pragma unroll
      for (int i = 0; i < 8; ++i) acc[mt][nt][i] = 0.0f;

  const bf16_t* aBase    = As + ((wm * 32 + ml) * LDS_RS + hi * 8);
  const int     nColBase = blockIdx.y * 64 + wn * 32 + ml;

  for (int kh = 0; kh < Kh; ++kh) {
    for (int kw = 0; kw < Kw; ++kw) {
      // per-tap, per-pixel row offsets (hoisted out of the cin loop)
      long long off[4]; bool ok[4];
#pragma unroll
      for (int e = 0; e < 4; ++e) {
        int ih = pm_oh[e] - ph + kh;
        int iw = pm_ow[e] - pw + kw;
        ok[e]  = pm_v[e] && (unsigned)ih < (unsigned)H && (unsigned)iw < (unsigned)W;
        off[e] = pm_base[e] + (long long)ih * W + iw;
      }
      const bf16_t* wtap = wp + (long long)(kh * Kw + kw) * KB * CoutP * 32;

      for (int kb = 0; kb < KB; ++kb) {
        const int c0 = kb * 32;
        // keep the weight stream ahead in GL2 (global_prefetch_b8)
        if (kb + 1 < KB)
          __builtin_prefetch(wtap + (((long long)(kb + 1) * CoutP + nColBase) * 32), 0, 1);

        __syncthreads();
#pragma unroll
        for (int e = 0; e < 4; ++e) {
          v4bf pk;
#pragma unroll
          for (int kk = 0; kk < 4; ++kk) {
            float v = 0.0f;
            if (ok[e]) {
              v = x[off[e] + (long long)(c0 + k4 * 4 + kk) * HW];
              if (reluIn) v = fmaxf(v, 0.0f);
            }
            pk[kk] = (bf16_t)v;
          }
          *(v4bf*)(As + (mr + e * 16) * LDS_RS + k4 * 4) = pk;   // ds_store_b64
        }
        __syncthreads();

        // A fragments: ISA 16-bit 16x32 layout; padded [m][k] -> 2x ds_load_b128
        v16bf a[2], b[2];
#pragma unroll
        for (int mt = 0; mt < 2; ++mt) {
          const v8bf* ap = (const v8bf*)(aBase + mt * 16 * LDS_RS);
          v8bf lo = ap[0];          // K = hi*8 .. +7
          v8bf hv = ap[2];          // K = 16 + hi*8 .. +7
#pragma unroll
          for (int j = 0; j < 8; ++j) { a[mt][j] = lo[j]; a[mt][j + 8] = hv[j]; }
        }
        // B fragments: K = hi*8..+7 and 16+hi*8..+7 per lane in packed weights
#pragma unroll
        for (int nt = 0; nt < 2; ++nt) {
          const v8bf* bv = (const v8bf*)(wtap +
              (((long long)kb * CoutP + (nColBase + nt * 16)) * 32 + hi * 8));
          v8bf lo = bv[0];
          v8bf hv = bv[2];
#pragma unroll
          for (int j = 0; j < 8; ++j) { b[nt][j] = lo[j]; b[nt][j + 8] = hv[j]; }
        }
#pragma unroll
        for (int mt = 0; mt < 2; ++mt)
#pragma unroll
          for (int nt = 0; nt < 2; ++nt)
            acc[mt][nt] = __builtin_amdgcn_wmma_f32_16x16x32_bf16(
                false, a[mt], false, b[nt], (short)0, acc[mt][nt], false, false);
      }
    }
  }

  // store: D layout m = r + (lane>=16)*8, n = lane&15
#pragma unroll
  for (int nt = 0; nt < 2; ++nt) {
    int nc = nColBase + nt * 16;
    if (nc < Cout) {
      const float bb = bias[nc];
#pragma unroll
      for (int mt = 0; mt < 2; ++mt) {
#pragma unroll
        for (int r = 0; r < 8; ++r) {
          int mg = m0 + wm * 32 + mt * 16 + r + hi * 8;
          if (mg < M) {
            int n  = mg / HoWo;
            int rr = mg - n * HoWo;
            int oh = rr / Wo;
            int ow = rr - oh * Wo;
            float v = acc[mt][nt][r] + bb;
            if (reluOut) v = fmaxf(v, 0.0f);
            y[((long long)(n * Cout + nc) * Yh + (oh * rstep + roff)) * Yw
              + (ow * cstep + coff)] = v;
          }
        }
      }
    }
  }
}

// ---------------------------------------------------------------------------
static inline void launch_conv(hipStream_t s,
    const float* x, const bf16_t* wp, const float* bias, float* y,
    int N, int Cin, int H, int W, int Cout, int Kh, int Kw, int ph, int pw,
    int Yh, int Yw, int rstep, int roff, int cstep, int coff,
    int reluIn, int reluOut)
{
  int Ho = H + 2 * ph - Kh + 1;
  int Wo = W + 2 * pw - Kw + 1;
  long long M = (long long)N * Ho * Wo;
  int CoutP = ((Cout + 63) / 64) * 64;
  dim3 grid((unsigned)((M + 63) / 64), (unsigned)(CoutP / 64));
  conv_wmma_kernel<<<grid, dim3(128), 0, s>>>(x, wp, bias, y,
      N, Cin, H, W, Cout, CoutP, Kh, Kw, ph, pw, Ho, Wo,
      Yh, Yw, rstep, roff, cstep, coff, reluIn, reluOut);
}

extern "C" void kernel_launch(void* const* d_in, const int* in_sizes, int n_in,
                              void* d_out, int out_size, void* d_ws, size_t ws_size,
                              hipStream_t stream)
{
  (void)n_in; (void)out_size; (void)ws_size;
  const float* X;
  const float *c1w, *c1b, *c2w, *c2b;
  const float *uw[4][5], *ub[4][5];

  // Detect flattening order: dict-insertion (x first) vs sorted keys (x last).
  const bool ins = (in_sizes[0] == 8 * 2048 * 16 * 16);
  if (ins) {
    X   = (const float*)d_in[0];
    c1w = (const float*)d_in[1]; c1b = (const float*)d_in[2];
    int idx = 3;
    for (int u = 0; u < 4; ++u)
      for (int j = 0; j < 5; ++j) {
        uw[u][j] = (const float*)d_in[idx++];
        ub[u][j] = (const float*)d_in[idx++];
      }
    c2w = (const float*)d_in[43]; c2b = (const float*)d_in[44];
  } else {
    c1b = (const float*)d_in[0]; c1w = (const float*)d_in[1];
    c2b = (const float*)d_in[2]; c2w = (const float*)d_in[3];
    int idx = 4;
    for (int u = 0; u < 4; ++u) {
      for (int j = 0; j < 5; ++j) ub[u][j] = (const float*)d_in[idx++];
      for (int j = 0; j < 5; ++j) uw[u][j] = (const float*)d_in[idx++];
    }
    X = (const float*)d_in[44];
  }

  // Workspace: three fp32 ping-pong activation buffers, then packed weights.
  const long long BUF = 33554432LL;  // 8*64*256*256
  float* b0 = (float*)d_ws;
  float* b1 = b0 + BUF;
  float* b2 = b1 + BUF;
  bf16_t* packBase = (bf16_t*)(b2 + BUF);
  long long packOff = 0;

  auto pack = [&](const float* w, int Cout, int Cin, int Kh, int Kw) -> const bf16_t* {
    int CoutP = ((Cout + 63) / 64) * 64;
    bf16_t* dst = packBase + packOff;
    long long tot = (long long)Kh * Kw * (Cin >> 5) * CoutP * 32;
    pack_w_kernel<<<(unsigned)((tot + 255) / 256), 256, 0, stream>>>(
        w, dst, Cout, CoutP, Cin, Kh, Kw, tot);
    packOff += tot;
    return dst;
  };

  // {Kh, Kw, ph, pw} for branches 1..4 and the residual conv (index 4)
  static const int KHW[5][4] = { {3,3,1,1}, {2,3,0,1}, {3,2,1,0}, {2,2,0,0}, {3,3,1,1} };

  const bf16_t* p_c1 = pack(c1w, 1024, 2048, 1, 1);
  const bf16_t* p_u[4][5];
  for (int u = 0; u < 4; ++u) {
    int ci = 1024 >> u, co = 512 >> u;
    for (int j = 0; j < 5; ++j) {
      int cin = (j == 4) ? co : ci;
      p_u[u][j] = pack(uw[u][j], co, cin, KHW[j][0], KHW[j][1]);
    }
  }
  const bf16_t* p_c2 = pack(c2w, 1, 64, 3, 3);

  // conv1: 1x1, 2048 -> 1024 @ 16x16
  launch_conv(stream, X, p_c1, c1b, b0, 8, 2048, 16, 16, 1024, 1, 1, 0, 0,
              16, 16, 1, 0, 1, 0, 0, 0);

  float* cur = b0;
  float* spare = b1;
  for (int u = 0; u < 4; ++u) {
    int ci = 1024 >> u, co = 512 >> u, S = 16 << u;
    float* inter = spare;
    long long isz = (long long)8 * co * (2 * S) * (2 * S);
    // exact zeros at the pad-missing interleave positions
    fill_zero_kernel<<<(unsigned)((isz + 255) / 256), 256, 0, stream>>>(inter, isz);
    // 4 branch convs write pixel-interleaved (stride 2) into `inter`
    for (int j = 0; j < 4; ++j) {
      int roff = (j >= 2) ? 1 : 0;
      int coff = (j & 1) ? 1 : 0;
      launch_conv(stream, cur, p_u[u][j], ub[u][j], inter, 8, ci, S, S, co,
                  KHW[j][0], KHW[j][1], KHW[j][2], KHW[j][3],
                  2 * S, 2 * S, 2, roff, 2, coff, 0, 0);
    }
    // t = conv3x3(relu(inter)) -> b2
    launch_conv(stream, inter, p_u[u][4], ub[u][4], b2, 8, co, 2 * S, 2 * S, co,
                3, 3, 1, 1, 2 * S, 2 * S, 1, 0, 1, 0, 1, 0);
    // inter = relu(inter + t)   (outer relu(out) is idempotent)
    residual_relu_kernel<<<(unsigned)((isz + 255) / 256), 256, 0, stream>>>(inter, b2, isz);
    spare = cur;
    cur = inter;
  }

  // conv2: 3x3, 64 -> 1 @ 256x256, relu on store -> b2
  launch_conv(stream, cur, p_c2, c2b, b2, 8, 64, 256, 256, 1, 3, 3, 1, 1,
              256, 256, 1, 0, 1, 0, 0, 1);

  // nearest 2x upsample -> d_out (8,1,512,512)
  long long tot = (long long)8 * 512 * 512;
  upsample2x_kernel<<<(unsigned)((tot + 255) / 256), 256, 0, stream>>>(
      b2, (float*)d_out, 8, 256, 256);
}
